// SutraV04_4501125726903
// MI455X (gfx1250) — matrix operand order repaired and checked by reading, not compiled, not run on hardware
//
#include <hip/hip_runtime.h>
#include <math.h>

// ---- model constants ----
#define DD    256
#define NP    1024
#define NB    16
#define BN    (NB*NP)      // 16384 patch rows
#define BTOK  (NB*4096)    // 65536 token rows
#define TOPK  8

typedef __attribute__((ext_vector_type(16))) _Float16 v16h;
typedef __attribute__((ext_vector_type(8)))  float    v8f;

__device__ __forceinline__ float gelu_exact(float x){
    return 0.5f * x * (1.0f + erff(x * 0.7071067811865475f));
}
__device__ __forceinline__ float sigmoid_(float x){ return 1.0f/(1.0f+expf(-x)); }

__device__ __forceinline__ v16h cvt16(float4 a, float4 b, float4 c, float4 d){
    v16h r;
    r[0]=(_Float16)a.x; r[1]=(_Float16)a.y; r[2]=(_Float16)a.z; r[3]=(_Float16)a.w;
    r[4]=(_Float16)b.x; r[5]=(_Float16)b.y; r[6]=(_Float16)b.z; r[7]=(_Float16)b.w;
    r[8]=(_Float16)c.x; r[9]=(_Float16)c.y; r[10]=(_Float16)c.z; r[11]=(_Float16)c.w;
    r[12]=(_Float16)d.x; r[13]=(_Float16)d.y; r[14]=(_Float16)d.z; r[15]=(_Float16)d.w;
    return r;
}

// ---------------------------------------------------------------------------
// WMMA GEMM, 16x64 per wave (4 accumulators share one A fragment):
//   C[M,N] = act( (accum?C:0) + A_f32[M,K](lda) @ W + bias )
// W pre-transposed f16, BT[n*ldbt + k]: B-fragment = one contiguous 32B load.
// 4 waves / block. nblk = N/64. nt=1 -> non-temporal stores (streamed output).
// ---------------------------------------------------------------------------
__global__ __launch_bounds__(128)
void gemm_wmma_k(const float* __restrict__ A, int lda,
                 const _Float16* __restrict__ BT, int ldbt,
                 const float* __restrict__ bias,
                 float* __restrict__ C, int ldc,
                 int nblk, int K, int act, int accum, int nt)
{
    int lane = threadIdx.x & 31;
    int tile = blockIdx.x * 4 + (threadIdx.x >> 5);
    int m0 = (tile / nblk) << 4;
    int n0 = (tile % nblk) << 6;          // 64-wide N block
    bool lo = lane < 16;
    int klo   = lo ? 0 : 16;              // A fragment K-subblock
    int khalf = lo ? 0 : 16;              // B fragment K-half

    const float* arow = A + (size_t)(m0 + (lane & 15)) * lda + klo;
    const _Float16* brow0 = BT + (size_t)(n0      + (lane & 15)) * ldbt + khalf;
    const _Float16* brow1 = BT + (size_t)(n0 + 16 + (lane & 15)) * ldbt + khalf;
    const _Float16* brow2 = BT + (size_t)(n0 + 32 + (lane & 15)) * ldbt + khalf;
    const _Float16* brow3 = BT + (size_t)(n0 + 48 + (lane & 15)) * ldbt + khalf;

    v8f acc0 = {0.f,0.f,0.f,0.f,0.f,0.f,0.f,0.f};
    v8f acc1 = acc0, acc2 = acc0, acc3 = acc0;

    for (int k0 = 0; k0 < K; k0 += 32) {
        __builtin_prefetch(arow + k0 + 32, 0, 3);   // WGP-scope prefetch of next A K-block
        float4 f0 = *(const float4*)(arow + k0);
        float4 f1 = *(const float4*)(arow + k0 + 4);
        float4 f2 = *(const float4*)(arow + k0 + 8);
        float4 f3 = *(const float4*)(arow + k0 + 12);
        v16h av = cvt16(f0, f1, f2, f3);
        v16h b0 = *(const v16h*)(brow0 + k0);
        v16h b1 = *(const v16h*)(brow1 + k0);
        v16h b2 = *(const v16h*)(brow2 + k0);
        v16h b3 = *(const v16h*)(brow3 + k0);
        acc0 = __builtin_amdgcn_wmma_f32_16x16x32_f16(false, av, false, b0, (short)0, acc0, false, false);
        acc1 = __builtin_amdgcn_wmma_f32_16x16x32_f16(false, av, false, b1, (short)0, acc1, false, false);
        acc2 = __builtin_amdgcn_wmma_f32_16x16x32_f16(false, av, false, b2, (short)0, acc2, false, false);
        acc3 = __builtin_amdgcn_wmma_f32_16x16x32_f16(false, av, false, b3, (short)0, acc3, false, false);
    }

    int rbase = m0 + (lo ? 0 : 8);
    v8f accs[4] = {acc0, acc1, acc2, acc3};
    for (int j = 0; j < 4; ++j) {
        int col = n0 + 16*j + (lane & 15);
        float badd = bias ? bias[col] : 0.f;
        for (int r = 0; r < 8; ++r) {
            size_t off = (size_t)(rbase + r) * ldc + col;
            float v = accs[j][r];
            if (accum) v += C[off];
            v += badd;
            if (act == 1) v = gelu_exact(v);
            if (nt) __builtin_nontemporal_store(v, &C[off]);
            else    C[off] = v;
        }
    }
}

// ---------------------------------------------------------------------------
// Batched WMMA attention scores, 16 queries x 64 keys per wave:
//   S[b][n][m] = (Q[b,n,:]·K[b,m,:]) / 16, causal mask fused in store.
// grid (256, 16) x 128: per b, 64 n-tiles x 16 m-blocks = 1024 wave tiles.
// ---------------------------------------------------------------------------
__global__ __launch_bounds__(128)
void scores_wmma_k(const float* __restrict__ Q, const float* __restrict__ Kb,
                   float* __restrict__ S)
{
    int b = blockIdx.y;
    int lane = threadIdx.x & 31;
    int tile = blockIdx.x * 4 + (threadIdx.x >> 5);
    int n0 = (tile >> 4) << 4;   // query tile (64 of them)
    int m0 = (tile & 15) << 6;   // key block of 64
    bool lo = lane < 16;
    int klo   = lo ? 0 : 16;
    int khalf = lo ? 0 : 16;

    const float* qp  = Q  + ((size_t)b*NP + n0      + (lane & 15))*DD + klo;
    const float* kp0 = Kb + ((size_t)b*NP + m0      + (lane & 15))*DD + khalf;
    const float* kp1 = Kb + ((size_t)b*NP + m0 + 16 + (lane & 15))*DD + khalf;
    const float* kp2 = Kb + ((size_t)b*NP + m0 + 32 + (lane & 15))*DD + khalf;
    const float* kp3 = Kb + ((size_t)b*NP + m0 + 48 + (lane & 15))*DD + khalf;

    v8f acc0 = {0.f,0.f,0.f,0.f,0.f,0.f,0.f,0.f};
    v8f acc1 = acc0, acc2 = acc0, acc3 = acc0;

    for (int k0 = 0; k0 < DD; k0 += 32) {
        float4 a0 = *(const float4*)(qp + k0);
        float4 a1 = *(const float4*)(qp + k0 + 4);
        float4 a2 = *(const float4*)(qp + k0 + 8);
        float4 a3 = *(const float4*)(qp + k0 + 12);
        v16h av = cvt16(a0, a1, a2, a3);
        #define LOADB(p) cvt16(*(const float4*)((p)+k0), *(const float4*)((p)+k0+4), \
                               *(const float4*)((p)+k0+8), *(const float4*)((p)+k0+12))
        v16h b0 = LOADB(kp0);
        v16h b1 = LOADB(kp1);
        v16h b2 = LOADB(kp2);
        v16h b3 = LOADB(kp3);
        #undef LOADB
        acc0 = __builtin_amdgcn_wmma_f32_16x16x32_f16(false, av, false, b0, (short)0, acc0, false, false);
        acc1 = __builtin_amdgcn_wmma_f32_16x16x32_f16(false, av, false, b1, (short)0, acc1, false, false);
        acc2 = __builtin_amdgcn_wmma_f32_16x16x32_f16(false, av, false, b2, (short)0, acc2, false, false);
        acc3 = __builtin_amdgcn_wmma_f32_16x16x32_f16(false, av, false, b3, (short)0, acc3, false, false);
    }

    int rbase = n0 + (lo ? 0 : 8);
    float* srow = S + (size_t)b*NP*NP;
    v8f accs[4] = {acc0, acc1, acc2, acc3};
    for (int j = 0; j < 4; ++j) {
        int colm = m0 + 16*j + (lane & 15);
        for (int r = 0; r < 8; ++r) {
            int rown = rbase + r;
            float v = accs[j][r] * 0.0625f;        // 1/sqrt(256)
            if (colm > rown) v = -INFINITY;
            srow[(size_t)rown*NP + colm] = v;
        }
    }
}

// ---- helper / elementwise kernels --------------------------------------

__global__ void fill_k(float* p, float v, int n){
    int i = blockIdx.x*256 + threadIdx.x;
    if (i < n) p[i] = v;
}

// transpose+quantize weight: in W[K,N] f32 -> WT[n*K + k] f16
__global__ void tr_k(const float* __restrict__ W, _Float16* __restrict__ WT, int K, int N){
    int idx = blockIdx.x*256 + threadIdx.x;
    if (idx < K*N){
        int k = idx / N; int n = idx - k*N;
        WT[(size_t)n*K + k] = (_Float16)W[idx];
    }
}

__global__ __launch_bounds__(256)
void embed_k(const int* __restrict__ x, const float* __restrict__ emb,
             const float* __restrict__ pos, float* __restrict__ out)
{
    int r = blockIdx.x, d = threadIdx.x;
    int tok = x[r];
    out[(size_t)r*DD + d] = emb[(size_t)tok*DD + d] + pos[(r & 3)*DD + d];
}

__global__ __launch_bounds__(256)
void gru_gate_k(const float* __restrict__ gi, const float* __restrict__ gh,
                float* __restrict__ h, float* __restrict__ gout, int t)
{
    int bn = blockIdx.x, d = threadIdx.x;
    size_t o3 = (size_t)bn*768;
    float ir = gi[o3 + d], iz = gi[o3 + 256 + d], in_ = gi[o3 + 512 + d];
    float hr = gh[o3 + d], hz = gh[o3 + 256 + d], hn  = gh[o3 + 512 + d];
    float r  = sigmoid_(ir + hr);
    float z  = sigmoid_(iz + hz);
    float n_ = tanhf(in_ + r*hn);
    size_t oh = (size_t)bn*DD + d;
    float hv = (1.f - z)*n_ + z*h[oh];
    h[oh] = hv;
    gout[((size_t)bn*4 + t)*DD + d] = hv;
}

// LayerNorm over last dim (256). addMode: 0 none, 1 same-row add, 2 add[row/4]
__global__ __launch_bounds__(256)
void ln256_k(const float* __restrict__ X, const float* __restrict__ Add,
             const float* __restrict__ g, const float* __restrict__ bb,
             float* __restrict__ Out, int addMode)
{
    __shared__ float sm[256];
    int row = blockIdx.x, d = threadIdx.x;
    float x = X[(size_t)row*DD + d];
    if (addMode == 1)      x += Add[(size_t)row*DD + d];
    else if (addMode == 2) x += Add[(size_t)(row >> 2)*DD + d];
    sm[d] = x; __syncthreads();
    for (int s = 128; s > 0; s >>= 1){ if (d < s) sm[d] += sm[d+s]; __syncthreads(); }
    float mean = sm[0] * (1.f/256.f);
    __syncthreads();
    float dx = x - mean;
    sm[d] = dx*dx; __syncthreads();
    for (int s = 128; s > 0; s >>= 1){ if (d < s) sm[d] += sm[d+s]; __syncthreads(); }
    float var = sm[0] * (1.f/256.f);
    Out[(size_t)row*DD + d] = dx * rsqrtf(var + 1e-5f) * g[d] + bb[d];
}

__global__ __launch_bounds__(256)
void meanP_k(const float* __restrict__ L, float* __restrict__ out){
    int bn = blockIdx.x, d = threadIdx.x;
    size_t b4 = (size_t)bn*4*DD + d;
    out[(size_t)bn*DD + d] =
        0.25f*(L[b4] + L[b4 + DD] + L[b4 + 2*DD] + L[b4 + 3*DD]);
}

// msgs hidden combine: G[bn] = mean_w gelu(S1[bn] + N1[bn-w] + b1), in-place on S1
__global__ __launch_bounds__(256)
void msg_combine_k(float* __restrict__ S1, const float* __restrict__ N1,
                   const float* __restrict__ b1)
{
    int bn = blockIdx.x, d = threadIdx.x;
    int n = bn & (NP - 1);
    size_t o = (size_t)bn*DD + d;
    float s = S1[o] + b1[d];
    float acc = 0.f;
    for (int w = 0; w <= 4; ++w){
        float t = s;
        if (n - w >= 0) t += N1[o - (size_t)w*DD];
        acc += gelu_exact(t);
    }
    S1[o] = acc * 0.2f;
}

// halting prob per batch: mode 0 -> 0, 1 -> sigmoid(mean_n h @ halt_w + b), 2 -> 1
__global__ __launch_bounds__(256)
void halt_k(const float* __restrict__ h, const float* __restrict__ hw,
            const float* __restrict__ hb, float* __restrict__ hp, int mode)
{
    __shared__ float sm[256];
    int b = blockIdx.x, d = threadIdx.x;
    const float* base = h + (size_t)b*NP*DD + d;
    float s = 0.f;
    for (int n = 0; n < NP; ++n) s += base[(size_t)n*DD];
    s *= (1.f/(float)NP);
    sm[d] = s * hw[d]; __syncthreads();
    for (int st = 128; st > 0; st >>= 1){ if (d < st) sm[d] += sm[d+st]; __syncthreads(); }
    if (d == 0){
        float v;
        if (mode == 0)      v = 0.f;
        else if (mode == 2) v = 1.f;
        else                v = sigmoid_(sm[0] + hb[0]);
        hp[b] = v;
    }
}

__global__ __launch_bounds__(256)
void accum_out_k(const float* __restrict__ h, const float* __restrict__ hp,
                 const float* __restrict__ rem, float* __restrict__ outmp)
{
    int row = blockIdx.x, d = threadIdx.x;
    int b = row >> 10;
    size_t o = (size_t)row*DD + d;
    outmp[o] += rem[b]*hp[b]*h[o];
}

__global__ void stats_k(const float* __restrict__ hp, float* __restrict__ rem,
                        float* __restrict__ kl, float pg, float lpg)
{
    __shared__ float hps[16];
    int t = threadIdx.x;
    if (t < 16) hps[t] = hp[t];
    __syncthreads();
    if (t == 0){
        float m = 0.f;
        for (int i = 0; i < 16; ++i) m += hps[i];
        m *= (1.f/16.f);
        kl[0] += pg * (lpg - logf(m + 1e-8f));
    }
    if (t < 16) rem[t] = rem[t] * (1.f - hps[t]);
}

__global__ __launch_bounds__(256)
void topk_k(const float* __restrict__ S, float* __restrict__ tv, int* __restrict__ ti)
{
    int rn = blockIdx.x*256 + threadIdx.x;   // 0..BN-1
    const float* srow = S + (size_t)rn * NP;
    float v[TOPK]; int id[TOPK];
    for (int i = 0; i < TOPK; ++i){ v[i] = -INFINITY; id[i] = 0; }
    for (int m = 0; m < NP; ++m){
        float s = srow[m];
        if (s > v[TOPK-1]){
            int p = TOPK-1;
            while (p > 0 && s > v[p-1]){ v[p] = v[p-1]; id[p] = id[p-1]; --p; }
            v[p] = s; id[p] = m;
        }
    }
    for (int i = 0; i < TOPK; ++i){ tv[(size_t)rn*TOPK+i] = v[i]; ti[(size_t)rn*TOPK+i] = id[i]; }
}

__global__ __launch_bounds__(256)
void retrieve_k(const float* __restrict__ tv, const int* __restrict__ ti,
                const float* __restrict__ V, float* __restrict__ out)
{
    int rn = blockIdx.x, d = threadIdx.x;
    int b = rn >> 10;
    float vv[TOPK]; int id[TOPK];
    for (int i = 0; i < TOPK; ++i){ vv[i] = tv[(size_t)rn*TOPK+i]; id[i] = ti[(size_t)rn*TOPK+i]; }
    float mx = vv[0];                 // sorted descending
    float w[TOPK], ws = 0.f;
    for (int i = 0; i < TOPK; ++i){ w[i] = expf(vv[i] - mx); ws += w[i]; }
    float inv = 1.f/ws;
    float acc = 0.f;
    for (int i = 0; i < TOPK; ++i)
        acc += w[i]*inv * V[((size_t)(b << 10) + id[i])*DD + d];
    out[(size_t)rn*DD + d] = acc;
}

__global__ void klout_k(const float* __restrict__ kl, float* __restrict__ out){
    out[16777216] = kl[0] * 0.25f;   // /MAX_ROUNDS
}

// ---------------------------------------------------------------------------

static inline void launch_gemm(const float* A, int lda, const _Float16* BT, int ldbt,
                               const float* bias, float* C, int ldc,
                               int M, int N, int K, int act, int accum, int nt, hipStream_t s)
{
    int blocks = (M/16)*(N/64)/4;
    gemm_wmma_k<<<blocks, 128, 0, s>>>(A, lda, BT, ldbt, bias, C, ldc, N/64, K, act, accum, nt);
}
static inline void launch_tr(const float* W, _Float16* WT, int K, int N, hipStream_t s){
    int n = K*N;
    tr_k<<<(n + 255)/256, 256, 0, s>>>(W, WT, K, N);
}

extern "C" void kernel_launch(void* const* d_in, const int* in_sizes, int n_in,
                              void* d_out, int out_size, void* d_ws, size_t ws_size,
                              hipStream_t stream)
{
    (void)in_sizes; (void)n_in; (void)out_size; (void)ws_size;
    const int*   x       = (const int*)  d_in[0];
    const float* emb     = (const float*)d_in[1];
    const float* pos     = (const float*)d_in[2];
    const float* gru_wih = (const float*)d_in[3];
    const float* gru_whh = (const float*)d_in[4];
    const float* gru_bih = (const float*)d_in[5];
    const float* gru_bhh = (const float*)d_in[6];
    const float* ln_p_g  = (const float*)d_in[7];
    const float* ln_p_b  = (const float*)d_in[8];
    const float* sum_w   = (const float*)d_in[9];
    const float* sum_b   = (const float*)d_in[10];
    const float* msg_w1  = (const float*)d_in[11];
    const float* msg_b1  = (const float*)d_in[12];
    const float* msg_w2  = (const float*)d_in[13];
    const float* msg_b2  = (const float*)d_in[14];
    const float* upd_w1  = (const float*)d_in[15];
    const float* upd_b1  = (const float*)d_in[16];
    const float* upd_w2  = (const float*)d_in[17];
    const float* upd_b2  = (const float*)d_in[18];
    const float* halt_w  = (const float*)d_in[19];
    const float* halt_b  = (const float*)d_in[20];
    const float* ln_mp_g = (const float*)d_in[21];
    const float* ln_mp_b = (const float*)d_in[22];
    const float* q_w = (const float*)d_in[23];  const float* q_b = (const float*)d_in[24];
    const float* k_w = (const float*)d_in[25];  const float* k_b = (const float*)d_in[26];
    const float* v_w = (const float*)d_in[27];  const float* v_b = (const float*)d_in[28];
    const float* o_w = (const float*)d_in[29];  const float* o_b = (const float*)d_in[30];
    const float* bc_w = (const float*)d_in[31]; const float* bc_b = (const float*)d_in[32];
    const float* ln_f_g = (const float*)d_in[33];
    const float* ln_f_b = (const float*)d_in[34];
    const float* head_w = (const float*)d_in[35];
    float* out = (float*)d_out;

    // ---- workspace arena (floats) ----
    float* Wf = (float*)d_ws;
    const size_t o_R0    = 0;                       // 16777216 : xemb -> scores -> fin
    const size_t o_R1    = o_R0 + 16777216;         // 12582912 : gi -> q,k,v
    const size_t o_R2    = o_R1 + 12582912;         // 12582912 : gh -> retrieved -> broad
    const size_t o_local = o_R2 + 12582912;         // 16777216
    const size_t o_h     = o_local + 16777216;      // 4194304  : GRU hidden
    const size_t o_hmp   = o_h   + 4194304;
    const size_t o_outmp = o_hmp + 4194304;
    const size_t o_S1    = o_outmp + 4194304;
    const size_t o_N1    = o_S1  + 4194304;
    const size_t o_Hd    = o_N1  + 4194304;         // also reused as patch-mean buffer
    const size_t o_u2    = o_Hd  + 4194304;
    const size_t o_topv  = o_u2  + 4194304;         // 131072
    const size_t o_topi  = o_topv + 131072;         // 131072 (int)
    const size_t o_rem   = o_topi + 131072;         // 16
    const size_t o_hp    = o_rem + 16;              // 16
    const size_t o_kl    = o_hp + 16;               // 1
    const size_t o_wt    = ((o_kl + 1 + 31)/32)*32; // f16 weight region (32B aligned)

    float* xemb   = Wf + o_R0;
    float* scores = Wf + o_R0;
    float* fin    = Wf + o_R0;
    float* gi     = Wf + o_R1;
    float* qbuf   = Wf + o_R1;
    float* kbuf   = Wf + o_R1 + 4194304;
    float* vbuf   = Wf + o_R1 + 8388608;
    float* gh     = Wf + o_R2;
    float* retr   = Wf + o_R2;
    float* broadb = Wf + o_R2;
    float* local  = Wf + o_local;
    float* h      = Wf + o_h;
    float* hmp    = Wf + o_hmp;
    float* outmp  = Wf + o_outmp;
    float* S1     = Wf + o_S1;
    float* N1     = Wf + o_N1;
    float* Hd     = Wf + o_Hd;
    float* meanb  = Wf + o_Hd;
    float* u2     = Wf + o_u2;
    float* topv   = Wf + o_topv;
    int*   topi   = (int*)(Wf + o_topi);
    float* rem    = Wf + o_rem;
    float* hp     = Wf + o_hp;
    float* kl     = Wf + o_kl;
    _Float16* wt  = (_Float16*)(Wf + o_wt);

    // transposed f16 weights (halves offsets)
    _Float16* wihT   = wt;              // 768x256
    _Float16* whhT   = wt + 196608;     // 768x256
    _Float16* sumT   = wt + 393216;     // 256x256
    _Float16* msgW1T = wt + 458752;     // 256x512
    _Float16* msgW2T = wt + 589824;     // 256x256
    _Float16* updW1T = wt + 655360;     // 256x512
    _Float16* updW2T = wt + 786432;
    _Float16* qT     = wt + 851968;
    _Float16* kT     = wt + 917504;
    _Float16* vT     = wt + 983040;
    _Float16* oT     = wt + 1048576;
    _Float16* bcT    = wt + 1114112;
    _Float16* headT  = wt + 1179648;

    // ---- weight prep (f32 -> f16 transposed) ----
    launch_tr(gru_wih, wihT, 256, 768, stream);
    launch_tr(gru_whh, whhT, 256, 768, stream);
    launch_tr(sum_w,   sumT, 256, 256, stream);
    launch_tr(msg_w1,  msgW1T, 512, 256, stream);
    launch_tr(msg_w2,  msgW2T, 256, 256, stream);
    launch_tr(upd_w1,  updW1T, 512, 256, stream);
    launch_tr(upd_w2,  updW2T, 256, 256, stream);
    launch_tr(q_w,  qT,  256, 256, stream);
    launch_tr(k_w,  kT,  256, 256, stream);
    launch_tr(v_w,  vT,  256, 256, stream);
    launch_tr(o_w,  oT,  256, 256, stream);
    launch_tr(bc_w, bcT, 256, 256, stream);
    launch_tr(head_w, headT, 256, 256, stream);

    // ---- init state ----
    fill_k<<<(BN*DD + 255)/256, 256, 0, stream>>>(h, 0.f, BN*DD);
    fill_k<<<(BN*DD + 255)/256, 256, 0, stream>>>(outmp, 0.f, BN*DD);
    fill_k<<<1, 256, 0, stream>>>(rem, 1.f, 16);
    fill_k<<<1, 256, 0, stream>>>(kl, 0.f, 1);

    // ---- embedding + positional ----
    embed_k<<<BTOK, 256, 0, stream>>>(x, emb, pos, xemb);

    // ---- GRU over 4-token patches ----
    for (int t = 0; t < 4; ++t) {
        launch_gemm(xemb + t*DD, 4*DD, wihT, 256, gru_bih, gi, 768, BN, 768, 256, 0, 0, 0, stream);
        launch_gemm(h, DD,            whhT, 256, gru_bhh, gh, 768, BN, 768, 256, 0, 0, 0, stream);
        gru_gate_k<<<BN, 256, 0, stream>>>(gi, gh, h, local, t);
    }
    ln256_k<<<BTOK, 256, 0, stream>>>(local, nullptr, ln_p_g, ln_p_b, local, 0);

    // summaries = mean over patch then sum_w
    meanP_k<<<BN, 256, 0, stream>>>(local, meanb);
    launch_gemm(meanb, DD, sumT, 256, sum_b, hmp, DD, BN, 256, 256, 0, 0, 0, stream);

    // ---- message passing with ACT halting ----
    const float LP = 0.2f;
    for (int step = 0; step < 4; ++step) {
        // msgs: split-K of concat([self, neigh]) @ msg_w1
        launch_gemm(hmp, DD, msgW1T,       512, nullptr, S1, DD, BN, 256, 256, 0, 0, 0, stream);
        launch_gemm(hmp, DD, msgW1T + 256, 512, nullptr, N1, DD, BN, 256, 256, 0, 0, 0, stream);
        msg_combine_k<<<BN, 256, 0, stream>>>(S1, N1, msg_b1);           // G in S1
        launch_gemm(S1, DD, msgW2T, 256, msg_b2, N1, DD, BN, 256, 256, 0, 0, 0, stream); // msgs in N1
        // update MLP: gelu(h@U1a + msgs@U1b + b1) @ U2 + b2
        launch_gemm(hmp, DD, updW1T,       512, upd_b1, Hd, DD, BN, 256, 256, 0, 0, 0, stream);
        launch_gemm(N1,  DD, updW1T + 256, 512, nullptr, Hd, DD, BN, 256, 256, 1, 1, 0, stream);
        launch_gemm(Hd,  DD, updW2T, 256, upd_b2, u2, DD, BN, 256, 256, 0, 0, 0, stream);
        ln256_k<<<BN, 256, 0, stream>>>(hmp, u2, ln_mp_g, ln_mp_b, hmp, 1);
        // halting
        int mode = (step < 2) ? 0 : (step == 3 ? 2 : 1);
        halt_k<<<NB, 256, 0, stream>>>(hmp, halt_w, halt_b, hp, mode);
        accum_out_k<<<BN, 256, 0, stream>>>(hmp, hp, rem, outmp);
        float pg = LP * powf(1.f - LP, (float)step);
        stats_k<<<1, 32, 0, stream>>>(hp, rem, kl, pg, logf(pg));
    }

    // ---- sparse top-k causal retrieval ----
    launch_gemm(outmp, DD, qT, 256, q_b, qbuf, DD, BN, 256, 256, 0, 0, 0, stream);
    launch_gemm(outmp, DD, kT, 256, k_b, kbuf, DD, BN, 256, 256, 0, 0, 0, stream);
    launch_gemm(outmp, DD, vT, 256, v_b, vbuf, DD, BN, 256, 256, 0, 0, 0, stream);
    scores_wmma_k<<<dim3(256, NB), 128, 0, stream>>>(qbuf, kbuf, scores);
    topk_k<<<BN/256, 256, 0, stream>>>(scores, topv, topi);
    retrieve_k<<<BN, 256, 0, stream>>>(topv, topi, vbuf, retr);
    // combined = out_mp + retrieved @ o_w + o_b   (accumulate into out_mp)
    launch_gemm(retr, DD, oT, 256, o_b, outmp, DD, BN, 256, 256, 0, 1, 0, stream);
    // broad = combined @ bc_w + bc_b
    launch_gemm(outmp, DD, bcT, 256, bc_b, broadb, DD, BN, 256, 256, 0, 0, 0, stream);

    // ---- broadcast, final LN, head ----
    ln256_k<<<BTOK, 256, 0, stream>>>(local, broadb, ln_f_g, ln_f_b, fin, 2);
    launch_gemm(fin, DD, headT, 256, nullptr, out, DD, BTOK, 256, 256, 0, 0, 1, stream);
    klout_k<<<1, 1, 0, stream>>>(kl, out);
}
